// LSTMLayer_45088566674130
// MI455X (gfx1250) — matrix operand order repaired and checked by reading, not compiled, not run on hardware
//
#include <hip/hip_runtime.h>
#include <hip/hip_bf16.h>
#include <math.h>

// LSTM layer for MI455X (gfx1250, wave32, WMMA).
// B=32, T=512, D=1024, H=1024, gates 4H=4096.
//
// Strategy:
//   - One-time prep kernels: x (f32->f16), combined weight [Wx;Wh] (f32 ->
//     f16, pre-swizzled into per-lane WMMA B-fragments in contiguous-k order),
//     state init (c from c0, h16 from h0).
//   - 512 stream-ordered step kernels. Each step computes
//       gates = [x_t ; h_{t-1}] @ [Wx ; Wh]   (M=32, K=2048, N=4096)
//     with v_wmma_f32_16x16x32_f16 (f32 accumulate), reduces partial tiles
//     via LDS float atomics, then applies bias + sigmoid/tanh cell update.
//   - h state is double-buffered in f16 (GEMM operand); c state f32 in ws;
//     f32 h_t streamed to d_out.

#define BB 32
#define TT 512
#define DD 1024
#define HH 1024
#define NG 4096   // 4*H

typedef __attribute__((ext_vector_type(16))) _Float16 v16h;
typedef __attribute__((ext_vector_type(8)))  _Float16 v8h;
typedef __attribute__((ext_vector_type(4)))  _Float16 v4h;
typedef __attribute__((ext_vector_type(8)))  float    v8f;

__device__ __forceinline__ v16h load_frag(const _Float16* p0, const _Float16* p1) {
  union { v16h v; v8h h[2]; } u;
  u.h[0] = *(const v8h*)p0;   // global_load_b128
  u.h[1] = *(const v8h*)p1;   // global_load_b128
  return u.v;
}

// ---------------------------------------------------------------- prep ----

// x [B,T,D] f32 -> f16, 4 elements/thread, exact grid.
__global__ __launch_bounds__(256)
void lstm_convert_x(const float* __restrict__ src, _Float16* __restrict__ dst) {
  size_t i = ((size_t)blockIdx.x * 256 + threadIdx.x) * 4;
  float4 v = *(const float4*)(src + i);
  v4h o;
  o.x = (_Float16)v.x; o.y = (_Float16)v.y;
  o.z = (_Float16)v.z; o.w = (_Float16)v.w;
  *(v4h*)(dst + i) = o;
}

// Build swizzled f16 combined weight:
//   w16s[(((ntile*64 + kc)*32 + lane)*16 + e]
// holds W[k][n] with (this code's contiguous-k B-fragment order, 32x16 f16):
//   n = ntile*16 + (lane&15)
//   k = kc*32 + (lane>=16 ? 16 : 0) + e
// where W = [Wx ; Wh] over k in [0,2048).
__global__ __launch_bounds__(256)
void lstm_build_w16s(const float* __restrict__ Wx, const float* __restrict__ Wh,
                     _Float16* __restrict__ w16s) {
  int o     = blockIdx.x * 256 + threadIdx.x;   // 0 .. 2048*4096-1
  int e     = o & 15;
  int lane  = (o >> 4) & 31;
  int kc    = (o >> 9) & 63;
  int ntile = o >> 15;                          // 0..255
  int k = kc * 32 + ((lane >= 16) ? 16 : 0) + e;
  int n = ntile * 16 + (lane & 15);
  float v = (k < DD) ? Wx[(size_t)k * NG + n]
                     : Wh[(size_t)(k - DD) * NG + n];
  w16s[o] = (_Float16)v;
}

__global__ __launch_bounds__(256)
void lstm_init_state(const float* __restrict__ c0, const float* __restrict__ h0,
                     float* __restrict__ c_ws, _Float16* __restrict__ h16) {
  int i = blockIdx.x * 256 + threadIdx.x;       // 0 .. B*H-1
  c_ws[i] = c0[i];
  h16[i]  = (_Float16)h0[i];
}

// ---------------------------------------------------------------- step ----

// Grid: 64 blocks (one per 16-wide slice of the H dimension), 256 threads
// (8 waves).  wave = gate(0..3) x K-half(0: x-part, 1: h-part).
// Each wave owns one 16(N)x1024->16 K-half slice for both M-tiles.
__global__ __launch_bounds__(256)
void lstm_step(const _Float16* __restrict__ x16,
               const _Float16* __restrict__ w16s,
               const float* __restrict__ bias,
               const _Float16* __restrict__ h_prev,
               _Float16* __restrict__ h_next,
               float* __restrict__ c_ws,
               float* __restrict__ out,
               int t) {
  __shared__ float gl[4][32][16];               // [gate][batch][jl], 8 KB

  const int tid  = threadIdx.x;
  const int lane = tid & 31;
  const int wave = tid >> 5;
  const int wg   = blockIdx.x;                  // j-slice 0..63

  // zero gate accumulator in LDS
  for (int i = tid; i < 4 * 32 * 16; i += 256) ((float*)gl)[i] = 0.0f;
  __syncthreads();

  const int g     = wave & 3;                   // gate i/f/g/o
  const int kh    = wave >> 2;                  // 0: k in [0,1024), 1: [1024,2048)
  const int ntile = wg + 64 * g;                // 0..255 over 4096 gate cols
  const int ln    = lane & 15;
  const int aoff  = (lane >= 16) ? 16 : 0;      // A-fragment K offset: k = 16*(lane>=16) + e

  v8f acc0 = {};                                // batch rows 0..15
  v8f acc1 = {};                                // batch rows 16..31

  // Branch-free, loop-invariant operand bases (kh is wave-uniform; select
  // once here so the K loop is pure load_b128 x6 -> wmma x2).
  const size_t    xrow = (size_t)TT * DD;       // per-batch stride in x16
  const _Float16* asrc = kh ? h_prev : (x16 + (size_t)t * DD);
  const size_t    arow = kh ? (size_t)HH : xrow;
  const _Float16* a0b  = asrc + (size_t)ln        * arow + aoff;
  const _Float16* a1b  = asrc + (size_t)(16 + ln) * arow + aoff;
  const _Float16* bb   =
      w16s + ((size_t)ntile * 64 * 32 + (size_t)lane) * 16
           + (size_t)(kh * 32) * (32 * 16);

#pragma unroll 4
  for (int kc = 0; kc < 32; ++kc) {
    const _Float16* bp = bb + (size_t)kc * (32 * 16);
    v16h bm = load_frag(bp, bp + 8);            // 16 contiguous halves / lane

    const _Float16* a0p = a0b + kc * 32;
    const _Float16* a1p = a1b + kc * 32;
    v16h a0 = load_frag(a0p, a0p + 8);
    v16h a1 = load_frag(a1p, a1p + 8);

    acc0 = __builtin_amdgcn_wmma_f32_16x16x32_f16(
        false, a0, false, bm, (short)0, acc0, false, false);
    acc1 = __builtin_amdgcn_wmma_f32_16x16x32_f16(
        false, a1, false, bm, (short)0, acc1, false, false);
  }

  // C/D layout: elem r -> (M = r + 8*(lane>=16), N = lane&15).
  // Two deterministic commutative adds per cell (x-half + h-half).
  const int mrow = (lane >= 16) ? 8 : 0;        // 8*(lane>=16)
#pragma unroll
  for (int r = 0; r < 8; ++r) {
    atomicAdd(&gl[g][r + mrow][ln], acc0[r]);
    atomicAdd(&gl[g][16 + r + mrow][ln], acc1[r]);
  }
  __syncthreads();

  // Elementwise cell update: 512 (b, jl) cells, 2 per thread.
  for (int e = tid; e < BB * 16; e += 256) {
    const int b  = e >> 4;
    const int jl = e & 15;
    const int n  = wg * 16 + jl;                // 0..1023

    float ip = gl[0][b][jl] + bias[n];
    float fp = gl[1][b][jl] + bias[HH + n];
    float gp = gl[2][b][jl] + bias[2 * HH + n];
    float op = gl[3][b][jl] + bias[3 * HH + n];

    float i_s = 1.0f / (1.0f + expf(-ip));
    float f_s = 1.0f / (1.0f + expf(-fp));
    float g_t = tanhf(gp);
    float o_s = 1.0f / (1.0f + expf(-op));

    const int si = b * HH + n;
    float cn = f_s * c_ws[si] + i_s * g_t;
    float hn = o_s * tanhf(cn);

    c_ws[si]   = cn;
    h_next[si] = (_Float16)hn;
    out[(size_t)b * TT * HH + (size_t)t * HH + n] = hn;
    if (t == TT - 1) {                          // tuple tail: (cT, hT)
      out[(size_t)BB * TT * HH + si]           = cn;
      out[(size_t)BB * TT * HH + BB * HH + si] = hn;
    }
  }
}

// -------------------------------------------------------------- launch ----

extern "C" void kernel_launch(void* const* d_in, const int* in_sizes, int n_in,
                              void* d_out, int out_size, void* d_ws, size_t ws_size,
                              hipStream_t stream) {
  const float* x    = (const float*)d_in[0];    // [B,T,D]
  const float* c0   = (const float*)d_in[1];    // [B,H]
  const float* h0   = (const float*)d_in[2];    // [B,H]
  const float* Wx   = (const float*)d_in[3];    // [D,4H]
  const float* Wh   = (const float*)d_in[4];    // [H,4H]
  const float* bias = (const float*)d_in[5];    // [4H]
  float* out = (float*)d_out;

  char* ws = (char*)d_ws;
  _Float16* x16  = (_Float16*)(ws);                       // 32 MB
  _Float16* w16s = (_Float16*)(ws + (size_t)33554432);    // 16 MB
  float*    c_ws = (float*)   (ws + (size_t)50331648);    // 128 KB
  _Float16* h16a = (_Float16*)(ws + (size_t)50462720);    // 64 KB
  _Float16* h16b = (_Float16*)(ws + (size_t)50528256);    // 64 KB

  lstm_convert_x<<<(BB * TT * DD) / (256 * 4), 256, 0, stream>>>(x, x16);
  lstm_build_w16s<<<((DD + HH) * NG) / 256, 256, 0, stream>>>(Wx, Wh, w16s);
  lstm_init_state<<<(BB * HH) / 256, 256, 0, stream>>>(c0, h0, c_ws, h16a);

  for (int t = 0; t < TT; ++t) {
    _Float16* hp = (t & 1) ? h16b : h16a;
    _Float16* hn = (t & 1) ? h16a : h16b;
    lstm_step<<<64, 256, 0, stream>>>(x16, w16s, bias, hp, hn, c_ws, out, t);
  }
}
